// SpatialConv_9577777070579
// MI455X (gfx1250) — matrix operand order repaired and checked by reading, not compiled, not run on hardware
//
#include <hip/hip_runtime.h>

// ---------------------------------------------------------------------------
// MI455X (gfx1250) GAT pipeline.
// GEMMs via v_wmma_f32_16x16x32_bf16 (wave32), segment softmax via 3-pass
// atomics. fp32 storage everywhere (pipeline is scatter/atomic bound).
// B-operand staged in LDS in WMMA fragment order so both A and B fragment
// gathers are 2x ds_load_b128 per 16x16 tile.
// ---------------------------------------------------------------------------

typedef __attribute__((ext_vector_type(16))) __bf16 bf16x16;
typedef __attribute__((ext_vector_type(8)))  float  f32x8;

union Frag { unsigned short u[16]; unsigned int w2[8]; bf16x16 v; };

__device__ __forceinline__ unsigned short f2bf(float f) {
    unsigned u = __float_as_uint(f);
    u += 0x7FFFu + ((u >> 16) & 1u);        // round-to-nearest-even
    return (unsigned short)(u >> 16);
}
__device__ __forceinline__ unsigned pk2bf(float lo, float hi) {
    return (unsigned)f2bf(lo) | ((unsigned)f2bf(hi) << 16);
}

// monotonic float <-> uint key for atomicMax on floats (incl. negatives)
__device__ __forceinline__ unsigned fkey(float f) {
    unsigned u = __float_as_uint(f);
    return (u & 0x80000000u) ? ~u : (u | 0x80000000u);
}
__device__ __forceinline__ float funkey(unsigned k) {
    unsigned u = (k & 0x80000000u) ? (k & 0x7FFFFFFFu) : ~k;
    return __uint_as_float(u);
}

#define HID 128
#define NHEAD 4
#define NEG_SLOPE 0.2f

// ---------------------------------------------------------------------------
// GEMM: out[M, ldo-cols] = act(A[M,K] @ W[K, ldw-cols] + bias)
// MODE 0: A dense row-major (lda == K)
// MODE 1: A row e = concat(x[src[e]], x[dst[e]], dfo[e])   (K == 384)
// Block: 256 threads = 8 waves; block tile = 128 rows x 128 cols
// (col offset = blockIdx.y*128). K staged via LDS in 32-wide bf16 chunks.
// sA: row-major (A frag = two contiguous 16B runs per lane)
// sWf: fragment order [tile][lane][elem] (B frag = one 32B run per lane)
// ---------------------------------------------------------------------------
template <int MODE, bool RELU>
__global__ void gemm_bf16_wmma(const float* __restrict__ A,
                               const float* __restrict__ W,
                               const float* __restrict__ bias,
                               float* __restrict__ out,
                               int M, int K, int ldw, int ldo,
                               const int* __restrict__ gsrc,
                               const int* __restrict__ gdst,
                               const float* __restrict__ gx,
                               const float* __restrict__ gdfo) {
    __shared__ __align__(16) unsigned short sA[128 * 32];     // rows x k
    __shared__ __align__(16) unsigned short sWf[8 * 32 * 16]; // [nt][lane][i]

    const int t    = threadIdx.x;
    const int lane = t & 31;
    const int w    = t >> 5;        // wave 0..7
    const int lm   = lane & 15;     // lane % 16
    const int lg   = lane >> 4;     // lane / 16
    const int rowBase = blockIdx.x * 128;
    const int n0      = blockIdx.y * 128;

    f32x8 acc[8];
#pragma unroll
    for (int i = 0; i < 8; ++i)
#pragma unroll
        for (int r = 0; r < 8; ++r) acc[i][r] = 0.0f;

    const int ar = t >> 1;          // A-load row 0..127
    const int ah = (t & 1) * 16;    // A-load k-half (0|16)
    const int wr = t >> 3;          // W-load k-row 0..31
    const int wnt = t & 7;          // W-load tile 0..7 (16 cols)
    // W swizzle destination: lane = j + 16*(wr>=16), elem = wr&15
    const int wlanehi = (wr >> 4) << 4;
    const int welem   = wr & 15;

    for (int kk = 0; kk < K; kk += 32) {
        // ---- stage A chunk (128 x 32), packed b128 stores ----
        {
            int grow = rowBase + ar;
            if (grow >= M) grow = M - 1;     // clamp; stores masked later
            float v[16];
            if (MODE == 0) {
                const float4* p4 =
                    reinterpret_cast<const float4*>(A + (size_t)grow * K + kk + ah);
                float4 qa = p4[0], qb = p4[1], qc = p4[2], qd = p4[3];
                v[0]=qa.x; v[1]=qa.y; v[2]=qa.z;  v[3]=qa.w;
                v[4]=qb.x; v[5]=qb.y; v[6]=qb.z;  v[7]=qb.w;
                v[8]=qc.x; v[9]=qc.y; v[10]=qc.z; v[11]=qc.w;
                v[12]=qd.x; v[13]=qd.y; v[14]=qd.z; v[15]=qd.w;
            } else {
                const int s = gsrc[grow], d = gdst[grow];
#pragma unroll
                for (int j = 0; j < 16; ++j) {
                    const int k = kk + ah + j;
                    float x;
                    if (k < HID)            x = gx[(size_t)s * HID + k];
                    else if (k < 2 * HID)   x = gx[(size_t)d * HID + (k - HID)];
                    else                    x = gdfo[(size_t)grow * HID + (k - 2 * HID)];
                    v[j] = x;
                }
            }
            uint4* dp = reinterpret_cast<uint4*>(&sA[ar * 32 + ah]);
            dp[0] = make_uint4(pk2bf(v[0], v[1]),  pk2bf(v[2], v[3]),
                               pk2bf(v[4], v[5]),  pk2bf(v[6], v[7]));
            dp[1] = make_uint4(pk2bf(v[8], v[9]),  pk2bf(v[10], v[11]),
                               pk2bf(v[12], v[13]), pk2bf(v[14], v[15]));
        }
        // ---- stage W chunk (32 x 128) swizzled to fragment order ----
        {
            const float4* p4 = reinterpret_cast<const float4*>(
                W + (size_t)(kk + wr) * ldw + n0 + wnt * 16);
            float4 qa = p4[0], qb = p4[1], qc = p4[2], qd = p4[3];
            float v[16] = {qa.x,qa.y,qa.z,qa.w, qb.x,qb.y,qb.z,qb.w,
                           qc.x,qc.y,qc.z,qc.w, qd.x,qd.y,qd.z,qd.w};
#pragma unroll
            for (int j = 0; j < 16; ++j)
                sWf[wnt * 512 + (j + wlanehi) * 16 + welem] = f2bf(v[j]);
        }
        __syncthreads();

        // ---- A fragment: two contiguous 16B runs (2x ds_load_b128) ----
        Frag a;
        {
            const unsigned short* abase = &sA[(16 * w + lm) * 32 + 16 * lg];
            const uint4 a0 = *reinterpret_cast<const uint4*>(abase);      // k=16g..16g+7
            const uint4 a1 = *reinterpret_cast<const uint4*>(abase + 8);  // k=16g+8..
            a.w2[0]=a0.x; a.w2[1]=a0.y; a.w2[2]=a0.z; a.w2[3]=a0.w;
            a.w2[4]=a1.x; a.w2[5]=a1.y; a.w2[6]=a1.z; a.w2[7]=a1.w;
        }
        // ---- all 8 B fragments: 32B contiguous per lane (2x b128 each) ----
        Frag b[8];
#pragma unroll
        for (int nt = 0; nt < 8; ++nt) {
            const uint4* bp =
                reinterpret_cast<const uint4*>(&sWf[nt * 512 + lane * 16]);
            const uint4 q0 = bp[0], q1 = bp[1];
            b[nt].w2[0]=q0.x; b[nt].w2[1]=q0.y; b[nt].w2[2]=q0.z; b[nt].w2[3]=q0.w;
            b[nt].w2[4]=q1.x; b[nt].w2[5]=q1.y; b[nt].w2[6]=q1.z; b[nt].w2[7]=q1.w;
        }
        // ---- 8 back-to-back WMMAs (distinct D tiles; no D->A/B hazards) ----
#pragma unroll
        for (int nt = 0; nt < 8; ++nt)
            acc[nt] = __builtin_amdgcn_wmma_f32_16x16x32_bf16(
                false, a.v, false, b[nt].v, (short)0, acc[nt], false, false);
        __syncthreads();
    }

    // ---- epilogue: bias (+ optional relu), masked store ----
#pragma unroll
    for (int nt = 0; nt < 8; ++nt) {
        const int gc = n0 + nt * 16 + lm;
        const float bv = bias ? bias[gc] : 0.0f;
#pragma unroll
        for (int r = 0; r < 8; ++r) {
            const int gr = rowBase + 16 * w + r + 8 * lg;
            if (gr < M) {
                float vv = acc[nt][r] + bv;
                if (RELU) vv = vv > 0.0f ? vv : 0.0f;
                out[(size_t)gr * ldo + gc] = vv;
            }
        }
    }
}

// ---------------------------------------------------------------------------
// Per-node attention terms: as[n,h] = <h[n,h,:], attS[h,:]>, likewise ad.
// ---------------------------------------------------------------------------
__global__ void node_attn_terms(const float* __restrict__ h,
                                const float* __restrict__ attS,
                                const float* __restrict__ attD,
                                float* __restrict__ asb, float* __restrict__ adb,
                                int T) {
    const int id = blockIdx.x * blockDim.x + threadIdx.x;
    if (id >= T * NHEAD) return;
    const int n = id >> 2, hh = id & 3;
    const float* hp = h + (size_t)n * (NHEAD * HID) + hh * HID;
    const float* sp = attS + hh * HID;
    const float* dp = attD + hh * HID;
    float s = 0.f, d = 0.f;
    for (int k = 0; k < HID; ++k) { const float v = hp[k]; s += v * sp[k]; d += v * dp[k]; }
    asb[id] = s; adb[id] = d;
}

// edge list decode: i < nReal -> (sArr[i], dArr[i], aE), else self loop i-nReal
__device__ __forceinline__ void edge_sd(int i, int nReal,
                                        const int* sArr, const int* dArr,
                                        int& s, int& d) {
    if (i < nReal) { s = sArr[i]; d = dArr[i]; } else { s = d = i - nReal; }
}
__device__ __forceinline__ float edge_alpha(int i, int hh, int s, int d, int nReal,
                                            const float* asb, const float* adb,
                                            const float* aE, const float* aSelf) {
    float a = asb[s * NHEAD + hh] + adb[d * NHEAD + hh];
    if (i < nReal) { if (aE)    a += aE[(size_t)i * NHEAD + hh]; }
    else           { if (aSelf) a += aSelf[hh]; }
    return a > 0.f ? a : NEG_SLOPE * a;          // LeakyReLU
}

__global__ void seg_max_k(int Etot, int nReal,
                          const int* __restrict__ sArr, const int* __restrict__ dArr,
                          const float* __restrict__ asb, const float* __restrict__ adb,
                          const float* __restrict__ aE, const float* __restrict__ aSelf,
                          unsigned* __restrict__ mI) {
    const int id = blockIdx.x * blockDim.x + threadIdx.x;
    if (id >= Etot * NHEAD) return;
    const int i = id >> 2, hh = id & 3;
    int s, d; edge_sd(i, nReal, sArr, dArr, s, d);
    const float a = edge_alpha(i, hh, s, d, nReal, asb, adb, aE, aSelf);
    atomicMax(&mI[(size_t)d * NHEAD + hh], fkey(a));
}

__global__ void seg_sum_k(int Etot, int nReal,
                          const int* __restrict__ sArr, const int* __restrict__ dArr,
                          const float* __restrict__ asb, const float* __restrict__ adb,
                          const float* __restrict__ aE, const float* __restrict__ aSelf,
                          const unsigned* __restrict__ mI, float* __restrict__ den) {
    const int id = blockIdx.x * blockDim.x + threadIdx.x;
    if (id >= Etot * NHEAD) return;
    const int i = id >> 2, hh = id & 3;
    int s, d; edge_sd(i, nReal, sArr, dArr, s, d);
    const float a = edge_alpha(i, hh, s, d, nReal, asb, adb, aE, aSelf);
    const float ex = __expf(a - funkey(mI[(size_t)d * NHEAD + hh]));
    atomicAdd(&den[(size_t)d * NHEAD + hh], ex);
}

// weighted scatter: agg[d, :] += h[s, :] * w   (128 threads/edge, float4 each)
__global__ void seg_agg_k(int Etot, int nReal,
                          const int* __restrict__ sArr, const int* __restrict__ dArr,
                          const float* __restrict__ asb, const float* __restrict__ adb,
                          const float* __restrict__ aE, const float* __restrict__ aSelf,
                          const unsigned* __restrict__ mI, const float* __restrict__ den,
                          const float* __restrict__ h, float* __restrict__ agg) {
    const int id = blockIdx.x * blockDim.x + threadIdx.x;
    if (id >= Etot * 128) return;
    const int i = id >> 7, q = id & 127;
    const int c = q * 4, hh = q >> 5;           // head = c / 128
    int s, d; edge_sd(i, nReal, sArr, dArr, s, d);
    const float a  = edge_alpha(i, hh, s, d, nReal, asb, adb, aE, aSelf);
    const float ex = __expf(a - funkey(mI[(size_t)d * NHEAD + hh]));
    const float wg = ex / (den[(size_t)d * NHEAD + hh] + 1e-16f);
    const float4 hv =
        *reinterpret_cast<const float4*>(h + (size_t)s * (NHEAD * HID) + c);
    float* ap = agg + (size_t)d * (NHEAD * HID) + c;
    atomicAdd(ap + 0, hv.x * wg);
    atomicAdd(ap + 1, hv.y * wg);
    atomicAdd(ap + 2, hv.z * wg);
    atomicAdd(ap + 3, hv.w * wg);
}

// out[n,c] = relu( mean_h agg[n,h,c] + b1[c] + b2[c] )
__global__ void head_mean_relu(const float* __restrict__ agg,
                               const float* __restrict__ b1,
                               const float* __restrict__ b2,
                               float* __restrict__ outp, int T) {
    const int id = blockIdx.x * blockDim.x + threadIdx.x;
    if (id >= T * HID) return;
    const int n = id >> 7, c = id & 127;
    const float* ap = agg + (size_t)n * (NHEAD * HID) + c;
    float v = 0.25f * (ap[0] + ap[HID] + ap[2 * HID] + ap[3 * HID]) + b1[c] + b2[c];
    outp[id] = v > 0.f ? v : 0.f;
}

// out[n,c] = relu( mean_h (agg[n,h,c] + gb[h*128+c]) + lb[c] )
__global__ void final_out_k(const float* __restrict__ agg,
                            const float* __restrict__ gb,
                            const float* __restrict__ lb,
                            float* __restrict__ outp, int T) {
    const int id = blockIdx.x * blockDim.x + threadIdx.x;
    if (id >= T * HID) return;
    const int n = id >> 7, c = id & 127;
    const float* ap = agg + (size_t)n * (NHEAD * HID) + c;
    float v = 0.25f * ((ap[0]       + gb[c]) +
                       (ap[HID]     + gb[HID + c]) +
                       (ap[2 * HID] + gb[2 * HID + c]) +
                       (ap[3 * HID] + gb[3 * HID + c])) + lb[c];
    outp[id] = v > 0.f ? v : 0.f;
}

// Ve[c,h] = sum_k en_We[c, h*128+k] * attE[h,k]   (folds the e_emb GEMM)
__global__ void compute_Ve(const float* __restrict__ enWe,
                           const float* __restrict__ attE,
                           float* __restrict__ Ve) {
    const int id = blockIdx.x * blockDim.x + threadIdx.x;
    if (id >= HID * NHEAD) return;
    const int c = id >> 2, hh = id & 3;
    float s = 0.f;
    for (int k = 0; k < HID; ++k)
        s += enWe[(size_t)c * (NHEAD * HID) + hh * HID + k] * attE[hh * HID + k];
    Ve[id] = s;
}

// column mean of df: one block per column, block reduction
__global__ void col_mean_k(const float* __restrict__ df, int E,
                           float* __restrict__ meandf) {
    __shared__ float sb[256];
    const int c = blockIdx.x;
    float s = 0.f;
    for (int e = threadIdx.x; e < E; e += 256) s += df[(size_t)e * HID + c];
    sb[threadIdx.x] = s;
    __syncthreads();
    for (int st = 128; st > 0; st >>= 1) {
        if (threadIdx.x < st) sb[threadIdx.x] += sb[threadIdx.x + st];
        __syncthreads();
    }
    if (threadIdx.x == 0) meandf[c] = sb[0] / (float)E;
}

__global__ void self_alpha_k(const float* __restrict__ meandf,
                             const float* __restrict__ Ve,
                             float* __restrict__ aSelf) {
    const int hh = threadIdx.x;
    if (hh >= NHEAD) return;
    float s = 0.f;
    for (int c = 0; c < HID; ++c) s += meandf[c] * Ve[c * NHEAD + hh];
    aSelf[hh] = s;
}

__global__ void edge_ae_k(const float* __restrict__ df,
                          const float* __restrict__ Ve,
                          float* __restrict__ aE, int E) {
    const int id = blockIdx.x * blockDim.x + threadIdx.x;
    if (id >= E * NHEAD) return;
    const int e = id >> 2, hh = id & 3;
    float s = 0.f;
    for (int c = 0; c < HID; ++c) s += df[(size_t)e * HID + c] * Ve[c * NHEAD + hh];
    aE[id] = s;
}

__global__ void copy_f32_k(const float* __restrict__ a, float* __restrict__ b,
                           long long n) {
    const long long i = (long long)blockIdx.x * blockDim.x + threadIdx.x;
    if (i < n) b[i] = a[i];
}
__global__ void fill_f32_k(float* p, float v, long long n) {
    const long long i = (long long)blockIdx.x * blockDim.x + threadIdx.x;
    if (i < n) p[i] = v;
}
__global__ void fill_u32_k(unsigned* p, unsigned v, long long n) {
    const long long i = (long long)blockIdx.x * blockDim.x + threadIdx.x;
    if (i < n) p[i] = v;
}

static inline int gb(long long n) { return (int)((n + 255) / 256); }

extern "C" void kernel_launch(void* const* d_in, const int* in_sizes, int n_in,
                              void* d_out, int out_size, void* d_ws, size_t ws_size,
                              hipStream_t stream) {
    const float* x        = (const float*)d_in[0];
    const float* dist     = (const float*)d_in[1];
    const float* distOrd  = (const float*)d_in[2];
    const float* projW    = (const float*)d_in[3];
    const float* projB    = (const float*)d_in[4];
    const float* efcW     = (const float*)d_in[5];
    const float* efcB     = (const float*)d_in[6];
    const float* eeW      = (const float*)d_in[7];
    const float* eeAs     = (const float*)d_in[8];
    const float* eeAd     = (const float*)d_in[9];
    const float* eeGb     = (const float*)d_in[10];
    const float* eeLb     = (const float*)d_in[11];
    const float* enW      = (const float*)d_in[12];
    const float* enWe     = (const float*)d_in[13];
    const float* enAs     = (const float*)d_in[14];
    const float* enAd     = (const float*)d_in[15];
    const float* enAe     = (const float*)d_in[16];
    const float* enGb     = (const float*)d_in[17];
    const float* enLb     = (const float*)d_in[18];
    const int*   eidx     = (const int*)d_in[19];
    const int*   e2e      = (const int*)d_in[20];

    const int N  = in_sizes[0] / HID;
    const int E  = in_sizes[19] / 2;
    const int EE = in_sizes[20] / 2;
    const int T  = N + E;

    const int* src    = eidx;
    const int* dst    = eidx + E;
    const int* eeSrc  = e2e;
    const int* eeDst  = e2e + EE;

    // workspace layout (fp32)
    float* ws = (float*)d_ws;
    size_t o = 0;
    float*    ne     = ws + o; o += (size_t)T * HID;
    float*    hbuf   = ws + o; o += (size_t)T * NHEAD * HID;
    float*    agg    = ws + o; o += (size_t)T * NHEAD * HID;
    float*    asb    = ws + o; o += (size_t)T * NHEAD;
    float*    adb    = ws + o; o += (size_t)T * NHEAD;
    unsigned* mI     = (unsigned*)(ws + o); o += (size_t)T * NHEAD;
    float*    den    = ws + o; o += (size_t)T * NHEAD;
    float*    ebuf   = ws + o; o += (size_t)E * HID;     // dfo, then df
    float*    aE     = ws + o; o += (size_t)E * NHEAD;
    float*    ne2    = ws + o; o += (size_t)T * HID;
    float*    Ve     = ws + o; o += (size_t)HID * NHEAD;
    float*    meandf = ws + o; o += HID;
    float*    aSelf  = ws + o; o += NHEAD;

    const int gE = (E + 127) / 128, gT = (T + 127) / 128;
    const unsigned NEGINF_KEY = 0x00800000u;   // fkey(-FLT_MAX)

    // 1) dfo = dist_feat_order @ proj_W + proj_b
    gemm_bf16_wmma<0, false><<<dim3(gE, 1), 256, 0, stream>>>(
        distOrd, projW, projB, ebuf, E, 64, HID, HID,
        nullptr, nullptr, nullptr, nullptr);

    // 2) ne[0:N] = x
    copy_f32_k<<<gb((long long)N * HID), 256, 0, stream>>>(x, ne, (long long)N * HID);

    // 3) ne[N:T] = relu(concat(x[src],x[dst],dfo) @ edge_fc_W + b)
    gemm_bf16_wmma<1, true><<<dim3(gE, 1), 256, 0, stream>>>(
        nullptr, efcW, efcB, ne + (size_t)N * HID, E, 3 * HID, HID, HID,
        src, dst, x, ebuf);

    // 4) h = ne @ ee_W        [T, 512]
    gemm_bf16_wmma<0, false><<<dim3(gT, 4), 256, 0, stream>>>(
        ne, eeW, nullptr, hbuf, T, HID, NHEAD * HID, NHEAD * HID,
        nullptr, nullptr, nullptr, nullptr);

    // 5) ee attention terms + segment softmax + scatter
    node_attn_terms<<<gb((long long)T * NHEAD), 256, 0, stream>>>(
        hbuf, eeAs, eeAd, asb, adb, T);
    fill_u32_k<<<gb((long long)T * NHEAD), 256, 0, stream>>>(mI, NEGINF_KEY, (long long)T * NHEAD);
    fill_f32_k<<<gb((long long)T * NHEAD), 256, 0, stream>>>(den, 0.f, (long long)T * NHEAD);
    fill_f32_k<<<gb((long long)T * NHEAD * HID), 256, 0, stream>>>(agg, 0.f, (long long)T * NHEAD * HID);
    {
        const int Etot = EE + T;
        seg_max_k<<<gb((long long)Etot * NHEAD), 256, 0, stream>>>(
            Etot, EE, eeSrc, eeDst, asb, adb, nullptr, nullptr, mI);
        seg_sum_k<<<gb((long long)Etot * NHEAD), 256, 0, stream>>>(
            Etot, EE, eeSrc, eeDst, asb, adb, nullptr, nullptr, mI, den);
        seg_agg_k<<<gb((long long)Etot * 128), 256, 0, stream>>>(
            Etot, EE, eeSrc, eeDst, asb, adb, nullptr, nullptr, mI, den, hbuf, agg);
    }
    // 6) ne2 = relu(mean_h agg + ee_gat_bias + ee_layer_bias)
    head_mean_relu<<<gb((long long)T * HID), 256, 0, stream>>>(agg, eeGb, eeLb, ne2, T);

    // 7) df = dist_feat @ proj_W + proj_b; folded edge-attention a_e = df @ Ve
    gemm_bf16_wmma<0, false><<<dim3(gE, 1), 256, 0, stream>>>(
        dist, projW, projB, ebuf, E, 64, HID, HID,
        nullptr, nullptr, nullptr, nullptr);
    compute_Ve<<<2, 256, 0, stream>>>(enWe, enAe, Ve);
    col_mean_k<<<HID, 256, 0, stream>>>(ebuf, E, meandf);
    self_alpha_k<<<1, 32, 0, stream>>>(meandf, Ve, aSelf);
    edge_ae_k<<<gb((long long)E * NHEAD), 256, 0, stream>>>(ebuf, Ve, aE, E);

    // 8) h = ne2 @ en_W       [T, 512]
    gemm_bf16_wmma<0, false><<<dim3(gT, 4), 256, 0, stream>>>(
        ne2, enW, nullptr, hbuf, T, HID, NHEAD * HID, NHEAD * HID,
        nullptr, nullptr, nullptr, nullptr);

    // 9) en attention + segment softmax + scatter
    node_attn_terms<<<gb((long long)T * NHEAD), 256, 0, stream>>>(
        hbuf, enAs, enAd, asb, adb, T);
    fill_u32_k<<<gb((long long)T * NHEAD), 256, 0, stream>>>(mI, NEGINF_KEY, (long long)T * NHEAD);
    fill_f32_k<<<gb((long long)T * NHEAD), 256, 0, stream>>>(den, 0.f, (long long)T * NHEAD);
    fill_f32_k<<<gb((long long)T * NHEAD * HID), 256, 0, stream>>>(agg, 0.f, (long long)T * NHEAD * HID);
    {
        const int Etot = E + T;
        seg_max_k<<<gb((long long)Etot * NHEAD), 256, 0, stream>>>(
            Etot, E, src, dst, asb, adb, aE, aSelf, mI);
        seg_sum_k<<<gb((long long)Etot * NHEAD), 256, 0, stream>>>(
            Etot, E, src, dst, asb, adb, aE, aSelf, mI, den);
        seg_agg_k<<<gb((long long)Etot * 128), 256, 0, stream>>>(
            Etot, E, src, dst, asb, adb, aE, aSelf, mI, den, hbuf, agg);
    }

    // 10) out = relu(mean_h(agg + en_gat_bias) + en_layer_bias)
    final_out_k<<<gb((long long)T * HID), 256, 0, stream>>>(
        agg, enGb, enLb, (float*)d_out, T);
}
